// SimplifiedEnhancedActor_32461362823520
// MI455X (gfx1250) — compile-verified, matrix operand order checked by reading, not run on hardware
//
#include <hip/hip_runtime.h>
#include <hip/hip_bf16.h>

// ---- problem dims ----
#define BB  32
#define FSZ 4
#define FTZ 6
#define FDZ 6
#define NN  512
#define KT  8
#define DD  4
#define VV  12
#define HH  256
#define NHD 4
#define DHD 64
#define WW  8

typedef __attribute__((ext_vector_type(16))) _Float16 v16h;
typedef __attribute__((ext_vector_type(8)))  float    v8f;

__device__ __forceinline__ float sigmf(float x) { return 1.f / (1.f + expf(-x)); }

// block-wide sum over 256 threads (8 wave32), LDS red[9]
__device__ __forceinline__ float block_sum256(float x, float* red) {
#pragma unroll
    for (int off = 16; off > 0; off >>= 1) x += __shfl_xor(x, off, 32);
    int wave = threadIdx.x >> 5;
    if ((threadIdx.x & 31) == 0) red[wave] = x;
    __syncthreads();
    if (threadIdx.x == 0) {
        float s = 0.f;
#pragma unroll
        for (int w = 0; w < 8; ++w) s += red[w];
        red[8] = s;
    }
    __syncthreads();
    float s = red[8];
    __syncthreads();
    return s;
}

// ---------------- FiLM conditioning: gs,bs,gd,bd per batch ----------------
__global__ __launch_bounds__(HH) void k_film(
    const float* __restrict__ weights,
    const float* __restrict__ we1_w, const float* __restrict__ we1_b,
    const float* __restrict__ we2_w, const float* __restrict__ we2_b,
    const float* __restrict__ was_w, const float* __restrict__ was_b,
    const float* __restrict__ wad_w, const float* __restrict__ wad_b,
    float* __restrict__ film /* [B][4][H] */) {
    __shared__ float e1[HH], we[HH];
    int b = blockIdx.x, h = threadIdx.x;
    float w0 = weights[b * 2 + 0], w1 = weights[b * 2 + 1];
    e1[h] = fmaxf(we1_w[h * 2 + 0] * w0 + we1_w[h * 2 + 1] * w1 + we1_b[h], 0.f);
    __syncthreads();
    float acc = we2_b[h];
    for (int i = 0; i < HH; ++i) acc += e1[i] * we2_w[h * HH + i];
    we[h] = acc;
    __syncthreads();
    float gs = was_b[h], bs = was_b[HH + h], gd = wad_b[h], bd = wad_b[HH + h];
    for (int i = 0; i < HH; ++i) {
        float v = we[i];
        gs += v * was_w[h * HH + i];
        bs += v * was_w[(HH + h) * HH + i];
        gd += v * wad_w[h * HH + i];
        bd += v * wad_w[(HH + h) * HH + i];
    }
    float* fb = film + (size_t)b * 4 * HH;
    fb[h] = gs; fb[HH + h] = bs; fb[2 * HH + h] = gd; fb[3 * HH + h] = bd;
}

// ---------------- decoder: LSTM cell + GRU memory + readout (batch-uniform) ----------------
__global__ __launch_bounds__(HH) void k_decoder(
    const float* __restrict__ coord_w, const float* __restrict__ coord_b,
    const float* __restrict__ x0, const float* __restrict__ h0, const float* __restrict__ c0,
    const float* __restrict__ lstm_wih, const float* __restrict__ lstm_whh,
    const float* __restrict__ lstm_bih, const float* __restrict__ lstm_bhh,
    const float* __restrict__ gru_wih, const float* __restrict__ gru_whh,
    const float* __restrict__ gru_bih, const float* __restrict__ gru_bhh,
    const float* __restrict__ ro_w, const float* __restrict__ ro_b,
    const float* __restrict__ veh_in_w, const float* __restrict__ veh_in_b,
    float* __restrict__ dec /* [4][H]: h_t,c_t,h_enr,q_veh */,
    float* __restrict__ out_ht, float* __restrict__ out_ct, float* __restrict__ out_hmem) {
    __shared__ float din[HH], hprev[HH], ht[HH], hcur[HH], msum[HH];
    int h = threadIdx.x;
    din[h]   = coord_w[h * 2 + 0] * x0[0] + coord_w[h * 2 + 1] * x0[1] + coord_b[h];
    hprev[h] = h0[h];
    __syncthreads();
    // LSTM gates (torch order i,f,g,o)
    float gi = lstm_bih[h] + lstm_bhh[h];
    float gf = lstm_bih[HH + h] + lstm_bhh[HH + h];
    float gg = lstm_bih[2 * HH + h] + lstm_bhh[2 * HH + h];
    float go = lstm_bih[3 * HH + h] + lstm_bhh[3 * HH + h];
    for (int i = 0; i < HH; ++i) {
        float x = din[i], hp = hprev[i];
        gi += x * lstm_wih[h * HH + i]            + hp * lstm_whh[h * HH + i];
        gf += x * lstm_wih[(HH + h) * HH + i]     + hp * lstm_whh[(HH + h) * HH + i];
        gg += x * lstm_wih[(2 * HH + h) * HH + i] + hp * lstm_whh[(2 * HH + h) * HH + i];
        go += x * lstm_wih[(3 * HH + h) * HH + i] + hp * lstm_whh[(3 * HH + h) * HH + i];
    }
    float c2  = sigmf(gf) * c0[h] + sigmf(gi) * tanhf(gg);
    float htv = sigmf(go) * tanhf(c2);
    ht[h] = htv;
    hcur[h] = 0.f;
    __syncthreads();
    // gi for GRU step 7 (input = h_t); steps 0..6 input = 0 -> gi = bih
    float r7 = gru_bih[h], z7 = gru_bih[HH + h], n7 = gru_bih[2 * HH + h];
    for (int i = 0; i < HH; ++i) {
        float x = ht[i];
        r7 += x * gru_wih[h * HH + i];
        z7 += x * gru_wih[(HH + h) * HH + i];
        n7 += x * gru_wih[(2 * HH + h) * HH + i];
    }
    __syncthreads();
    float msumv = 0.f;
    for (int w = 0; w < WW; ++w) {
        float hr = gru_bhh[h], hz = gru_bhh[HH + h], hn = gru_bhh[2 * HH + h];
        for (int i = 0; i < HH; ++i) {
            float x = hcur[i];
            hr += x * gru_whh[h * HH + i];
            hz += x * gru_whh[(HH + h) * HH + i];
            hn += x * gru_whh[(2 * HH + h) * HH + i];
        }
        float ir  = (w == 7) ? r7 : gru_bih[h];
        float iz  = (w == 7) ? z7 : gru_bih[HH + h];
        float inn = (w == 7) ? n7 : gru_bih[2 * HH + h];
        float r = sigmf(ir + hr), z = sigmf(iz + hz);
        float n = tanhf(inn + r * hn);
        float h2 = (1.f - z) * n + z * hcur[h];
        __syncthreads();
        hcur[h] = h2;
        msumv += h2;
        __syncthreads();
    }
    msum[h] = msumv * (1.f / WW);
    __syncthreads();
    float mc = ro_b[h];
    for (int i = 0; i < HH; ++i) mc += msum[i] * ro_w[h * HH + i];
    float henr = htv + tanhf(mc);
    din[h] = henr;  // stage h_enr for q projection
    __syncthreads();
    float q = veh_in_b[h];
    for (int i = 0; i < HH; ++i) q += din[i] * veh_in_w[h * HH + i];
    dec[h] = htv; dec[HH + h] = c2; dec[2 * HH + h] = henr; dec[3 * HH + h] = q;
    // batch-broadcast outputs: h_t, c_t, h_mem_new (7 zero rows + h_t)
    for (int b = 0; b < BB; ++b) {
        out_ht[b * HH + h] = htv;
        out_ct[b * HH + h] = c2;
        for (int w = 0; w < WW - 1; ++w) out_hmem[((size_t)b * WW + w) * HH + h] = 0.f;
        out_hmem[((size_t)b * WW + WW - 1) * HH + h] = htv;
    }
}

// ---------------- static 1x1 conv + FiLM -> fp16 rows for WMMA ----------------
__global__ __launch_bounds__(HH) void k_sproj(
    const float* __restrict__ stat, const float* __restrict__ static_w,
    const float* __restrict__ static_b, const float* __restrict__ film,
    _Float16* __restrict__ s16) {
    int bn = blockIdx.x, b = bn >> 9, n = bn & (NN - 1), h = threadIdx.x;
    float acc = static_b[h];
#pragma unroll
    for (int f = 0; f < FSZ; ++f) acc += static_w[h * FSZ + f] * stat[((size_t)b * FSZ + f) * NN + n];
    const float* fb = film + (size_t)b * 4 * HH;
    acc = acc * (1.f + fb[h]) + fb[HH + h];
    s16[(size_t)bn * HH + h] = (_Float16)acc;
}

// ---------------- pack Ws (comb_w[:, :H]) and Wk (cust in_w k-rows) to fp16 ----------------
__global__ __launch_bounds__(HH) void k_pack(
    const float* __restrict__ comb_w, const float* __restrict__ cust_in_w,
    _Float16* __restrict__ Ws16, _Float16* __restrict__ Wk16) {
    int r = blockIdx.x, k = threadIdx.x;
    Ws16[r * HH + k] = (_Float16)comb_w[(size_t)r * 2 * HH + k];
    Wk16[r * HH + k] = (_Float16)cust_in_w[(size_t)(HH + r) * HH + k];
}

// ---------------- WMMA GEMM: C[M,Nc] = A[M,256] * W[Nc,256]^T (fp16 in, fp32 out) ----------
// one 16x16 tile per wave32, 8 waves/block, fragments gathered per ISA 16-bit layouts
__global__ __launch_bounds__(256) void k_gemm_f16(
    const _Float16* __restrict__ Amat, const _Float16* __restrict__ Wmat,
    float* __restrict__ Cmat, int Ncols) {
    int lane = threadIdx.x & 31;
    int wave = threadIdx.x >> 5;
    int tile = blockIdx.x * 8 + wave;
    int tilesN = Ncols >> 4;
    int tm = tile / tilesN, tn = tile % tilesN;
    int halfrow = lane & 15;
    int koff = (lane >> 4) * 8;          // lanes 16-31 hold K+8 / K+24 per ISA layout
    const _Float16* Arow = Amat + (size_t)(tm * 16 + halfrow) * HH + koff;
    const _Float16* Brow = Wmat + (size_t)(tn * 16 + halfrow) * HH + koff;
    v8f acc = {};
#pragma unroll
    for (int k0 = 0; k0 < HH; k0 += 32) {
        v16h a, bf;
#pragma unroll
        for (int j = 0; j < 8; ++j) {
            int kk = k0 + ((j & 3) * 2) + ((j >> 2) * 16);  // K-pair contiguous -> dword loads
            a[2 * j]      = Arow[kk];
            a[2 * j + 1]  = Arow[kk + 1];
            bf[2 * j]     = Brow[kk];
            bf[2 * j + 1] = Brow[kk + 1];
        }
        acc = __builtin_amdgcn_wmma_f32_16x16x32_f16(false, a, false, bf, (short)0, acc,
                                                     false, false);
    }
    // C/D layout: lanes 0-15 -> M rows 0..7, lanes 16-31 -> rows 8..15; N = lane&15
    int nOut = tn * 16 + halfrow;
    int mBase = tm * 16 + (lane >> 4) * 8;
#pragma unroll
    for (int r = 0; r < 8; ++r)
        Cmat[(size_t)(mBase + r) * Ncols + nOut] = acc[r];
}

// ---------------- vehicle branch of pairwise pre-activation: Cv = v_t @ Wv^T + comb_b ------
__global__ __launch_bounds__(HH) void k_cv(
    const float* __restrict__ trucks, const float* __restrict__ drones,
    const float* __restrict__ truck_w, const float* __restrict__ truck_b,
    const float* __restrict__ drone_w, const float* __restrict__ drone_b,
    const float* __restrict__ film, const float* __restrict__ comb_w,
    const float* __restrict__ comb_b, float* __restrict__ Cv) {
    __shared__ float vrow[HH];
    int bv = blockIdx.x, b = bv / VV, v = bv % VV, h = threadIdx.x;
    float acc;
    if (v < KT) {
        acc = truck_b[h];
#pragma unroll
        for (int f = 0; f < FTZ; ++f) acc += truck_w[h * FTZ + f] * trucks[((size_t)b * FTZ + f) * KT + v];
    } else {
        int dv = v - KT;
        acc = drone_b[h];
#pragma unroll
        for (int f = 0; f < FDZ; ++f) acc += drone_w[h * FDZ + f] * drones[((size_t)b * FDZ + f) * DD + dv];
    }
    const float* fb = film + (size_t)b * 4 * HH;
    vrow[h] = acc * (1.f + fb[2 * HH + h]) + fb[3 * HH + h];
    __syncthreads();
    float c = comb_b[h];
    for (int i = 0; i < HH; ++i) c += vrow[i] * comb_w[(size_t)h * 2 * HH + HH + i];
    Cv[(size_t)bv * HH + h] = c;
}

// ---------------- veh_repr[b,v] = mean_n LN(relu(A[b,n]+Cv[b,v])) ----------------
__global__ __launch_bounds__(HH) void k_vehrepr(
    const float* __restrict__ Amat, const float* __restrict__ Cv,
    const float* __restrict__ comb_g, const float* __restrict__ comb_bb,
    float* __restrict__ vrep) {
    __shared__ float red[9];
    int bv = blockIdx.x, b = bv / VV, h = threadIdx.x;
    float cvh = Cv[(size_t)bv * HH + h];
    float g = comb_g[h], bb = comb_bb[h];
    const float* Ab = Amat + (size_t)b * NN * HH;
    float acc = 0.f;
    for (int n = 0; n < NN; ++n) {
        float x = fmaxf(Ab[(size_t)n * HH + h] + cvh, 0.f);
        float mean = block_sum256(x, red) * (1.f / HH);
        float d = x - mean;
        float var = block_sum256(d * d, red) * (1.f / HH);
        acc += d * rsqrtf(var + 1e-5f) * g + bb;
    }
    vrep[(size_t)bv * HH + h] = acc * (1.f / NN);
}

// ---------------- vehicle pointer MHA + LN + probs + argmax ----------------
__global__ __launch_bounds__(HH) void k_vehattn(
    const float* __restrict__ dec, const float* __restrict__ vrep,
    const float* __restrict__ veh_in_w, const float* __restrict__ veh_in_b,
    const float* __restrict__ veh_out_w, const float* __restrict__ veh_out_b,
    const float* __restrict__ lnv_g, const float* __restrict__ lnv_b,
    const float* __restrict__ mask_veh,
    float* __restrict__ veh_out, int* __restrict__ veh_idx,
    float* __restrict__ out_vehprobs, float* __restrict__ out_vehidx) {
    __shared__ float kh[VV][HH], vh[VV][HH], attn[NHD][VV], stage[HH], outc[HH], vw[VV], red[9];
    int b = blockIdx.x, h = threadIdx.x;
    const float* q = dec + 3 * HH;  // precomputed h_enr @ Wq + bq
    for (int v = 0; v < VV; ++v) {
        stage[h] = vrep[(size_t)(b * VV + v) * HH + h];
        __syncthreads();
        float ka = veh_in_b[HH + h], va = veh_in_b[2 * HH + h];
        for (int i = 0; i < HH; ++i) {
            float x = stage[i];
            ka += x * veh_in_w[(size_t)(HH + h) * HH + i];
            va += x * veh_in_w[(size_t)(2 * HH + h) * HH + i];
        }
        kh[v][h] = ka; vh[v][h] = va;
        __syncthreads();
    }
    if (h < NHD * VV) {
        int head = h / VV, v = h % VV;
        float s = 0.f;
        for (int j = 0; j < DHD; ++j) s += q[head * DHD + j] * kh[v][head * DHD + j];
        s *= 0.125f;  // 1/sqrt(64)
        if (mask_veh[b * VV + v] < 0.5f) s = -1e9f;  // key_padding mask
        attn[head][v] = s;
    }
    __syncthreads();
    if (h < NHD) {  // softmax over V per head
        float mx = -1e30f;
        for (int v = 0; v < VV; ++v) mx = fmaxf(mx, attn[h][v]);
        float s = 0.f;
        for (int v = 0; v < VV; ++v) { float e = expf(attn[h][v] - mx); attn[h][v] = e; s += e; }
        float inv = 1.f / s;
        for (int v = 0; v < VV; ++v) attn[h][v] *= inv;
    }
    __syncthreads();
    {
        int head = h / DHD;
        float o = 0.f;
        for (int v = 0; v < VV; ++v) o += attn[head][v] * vh[v][h];
        outc[h] = o;
    }
    __syncthreads();
    float p = veh_out_b[h];
    for (int i = 0; i < HH; ++i) p += outc[i] * veh_out_w[h * HH + i];
    float mean = block_sum256(p, red) * (1.f / HH);
    float d = p - mean;
    float var = block_sum256(d * d, red) * (1.f / HH);
    veh_out[b * HH + h] = d * rsqrtf(var + 1e-5f) * lnv_g[h] + lnv_b[h];
    if (h < VV) {
        float w = 0.f;
        for (int head = 0; head < NHD; ++head) w += attn[head][h];
        w = w * (1.f / NHD) + 1e-8f;
        vw[h] = (mask_veh[b * VV + h] > 0.5f) ? w : 0.f;  // softmax(log(w+eps)) == renorm
    }
    __syncthreads();
    if (h == 0) {
        float s = 0.f;
        for (int v = 0; v < VV; ++v) s += vw[v];
        float inv = 1.f / s;
        int best = 0; float bp = -1.f;
        for (int v = 0; v < VV; ++v) {
            float pr = vw[v] * inv;
            out_vehprobs[b * VV + v] = pr;
            if (pr > bp) { bp = pr; best = v; }
        }
        veh_idx[b] = best;
        out_vehidx[b] = (float)best;
    }
}

// ---------------- sel rows: LN(relu(A + Cv[veh_idx])) -> fp16 for WMMA ----------------
__global__ __launch_bounds__(HH) void k_sel(
    const float* __restrict__ Amat, const float* __restrict__ Cv,
    const int* __restrict__ veh_idx,
    const float* __restrict__ comb_g, const float* __restrict__ comb_bb,
    _Float16* __restrict__ sel16) {
    __shared__ float red[9];
    int bn = blockIdx.x, b = bn >> 9, h = threadIdx.x;
    int v = veh_idx[b];
    float x = fmaxf(Amat[(size_t)bn * HH + h] + Cv[(size_t)(b * VV + v) * HH + h], 0.f);
    float mean = block_sum256(x, red) * (1.f / HH);
    float d = x - mean;
    float var = block_sum256(d * d, red) * (1.f / HH);
    sel16[(size_t)bn * HH + h] = (_Float16)(d * rsqrtf(var + 1e-5f) * comb_g[h] + comb_bb[h]);
}

// ---------------- customer pointer attention -> cust_probs (out-proj not needed) ----------
__global__ __launch_bounds__(HH) void k_custattn(
    const float* __restrict__ veh_out, const float* __restrict__ KH,
    const float* __restrict__ cust_in_w, const float* __restrict__ cust_in_b,
    const float* __restrict__ mask_pw, const int* __restrict__ veh_idx,
    float* __restrict__ out_custprobs) {
    __shared__ float q[HH], stage[HH], scores[NHD][NN], wbuf[NN], qbh[NHD], red[9];
    int b = blockIdx.x, h = threadIdx.x;
    int vsel = veh_idx[b];
    stage[h] = veh_out[b * HH + h];
    __syncthreads();
    float qv = cust_in_b[h];
    for (int i = 0; i < HH; ++i) qv += stage[i] * cust_in_w[h * HH + i];
    q[h] = qv;
    __syncthreads();
    if (h < NHD) {  // q . bk per head (bias folded analytically)
        float s = 0.f;
        for (int j = 0; j < DHD; ++j) s += q[h * DHD + j] * cust_in_b[HH + h * DHD + j];
        qbh[h] = s;
    }
    __syncthreads();
    int head = h >> 6, j0 = h & 63;
    for (int n = j0; n < NN; n += 64) {
        const float* krow = KH + ((size_t)b * NN + n) * HH + head * DHD;
        float s = 0.f;
        for (int j = 0; j < DHD; ++j) s += q[head * DHD + j] * krow[j];
        s = (s + qbh[head]) * 0.125f;
        if (mask_pw[((size_t)b * VV + vsel) * NN + n] < 0.5f) s = -1e9f;
        scores[head][n] = s;
    }
    __syncthreads();
    if (h < NHD) {  // softmax over N per head
        float mx = -1e30f;
        for (int n = 0; n < NN; ++n) mx = fmaxf(mx, scores[h][n]);
        float s = 0.f;
        for (int n = 0; n < NN; ++n) { float e = expf(scores[h][n] - mx); scores[h][n] = e; s += e; }
        float inv = 1.f / s;
        for (int n = 0; n < NN; ++n) scores[h][n] *= inv;
    }
    __syncthreads();
    float part = 0.f;
    for (int n = h; n < NN; n += HH) {
        float w = 0.f;
        for (int hd = 0; hd < NHD; ++hd) w += scores[hd][n];
        w = w * (1.f / NHD) + 1e-8f;
        if (mask_pw[((size_t)b * VV + vsel) * NN + n] < 0.5f) w = 0.f;
        wbuf[n] = w;
        part += w;
    }
    float tot = block_sum256(part, red);
    float inv = 1.f / tot;
    for (int n = h; n < NN; n += HH) out_custprobs[(size_t)b * NN + n] = wbuf[n] * inv;
}

extern "C" void kernel_launch(void* const* d_in, const int* in_sizes, int n_in,
                              void* d_out, int out_size, void* d_ws, size_t ws_size,
                              hipStream_t stream) {
    (void)in_sizes; (void)n_in; (void)out_size; (void)ws_size;
    // inputs
    const float* stat_   = (const float*)d_in[0];
    const float* trucks  = (const float*)d_in[1];
    const float* drones  = (const float*)d_in[2];
    const float* weights = (const float*)d_in[3];
    const float* mask_v  = (const float*)d_in[4];
    const float* mask_pw = (const float*)d_in[5];
    const float* static_w = (const float*)d_in[6];
    const float* static_b = (const float*)d_in[7];
    const float* truck_w  = (const float*)d_in[8];
    const float* truck_b  = (const float*)d_in[9];
    const float* drone_w  = (const float*)d_in[10];
    const float* drone_b  = (const float*)d_in[11];
    const float* we1_w = (const float*)d_in[12];
    const float* we1_b = (const float*)d_in[13];
    const float* we2_w = (const float*)d_in[14];
    const float* we2_b = (const float*)d_in[15];
    const float* was_w = (const float*)d_in[16];
    const float* was_b = (const float*)d_in[17];
    const float* wad_w = (const float*)d_in[18];
    const float* wad_b = (const float*)d_in[19];
    const float* comb_w  = (const float*)d_in[20];
    const float* comb_b  = (const float*)d_in[21];
    const float* comb_g  = (const float*)d_in[22];
    const float* comb_bb = (const float*)d_in[23];
    const float* lstm_wih = (const float*)d_in[24];
    const float* lstm_whh = (const float*)d_in[25];
    const float* lstm_bih = (const float*)d_in[26];
    const float* lstm_bhh = (const float*)d_in[27];
    const float* gru_wih = (const float*)d_in[28];
    const float* gru_whh = (const float*)d_in[29];
    const float* gru_bih = (const float*)d_in[30];
    const float* gru_bhh = (const float*)d_in[31];
    const float* ro_w = (const float*)d_in[32];
    const float* ro_b = (const float*)d_in[33];
    const float* veh_in_w  = (const float*)d_in[34];
    const float* veh_in_b  = (const float*)d_in[35];
    const float* veh_out_w = (const float*)d_in[36];
    const float* veh_out_b = (const float*)d_in[37];
    const float* cust_in_w = (const float*)d_in[38];
    const float* cust_in_b = (const float*)d_in[39];
    const float* lnv_g = (const float*)d_in[42];
    const float* lnv_b = (const float*)d_in[43];
    const float* coord_w = (const float*)d_in[46];
    const float* coord_b = (const float*)d_in[47];
    const float* x0p = (const float*)d_in[48];
    const float* h0p = (const float*)d_in[49];
    const float* c0p = (const float*)d_in[50];

    // outputs (flat, return order)
    float* out = (float*)d_out;
    float* out_vehprobs = out;                     // 384
    float* out_custprobs = out + 384;              // 16384
    float* out_vehidx = out + 16768;               // 32
    float* out_ht = out + 16800;                   // 8192
    float* out_ct = out + 24992;                   // 8192
    float* out_hmem = out + 33184;                 // 65536

    // workspace carve-up (floats)
    float* W = (float*)d_ws;
    size_t off = 0;
    float* film = W + off; off += (size_t)BB * 4 * HH;
    float* dec  = W + off; off += 4 * HH;
    float* Cv   = W + off; off += (size_t)BB * VV * HH;
    float* vrep = W + off; off += (size_t)BB * VV * HH;
    float* vout = W + off; off += (size_t)BB * HH;
    int*   vidx = (int*)(W + off); off += BB;
    float* Amat = W + off; off += (size_t)BB * NN * HH;
    float* KHm  = W + off; off += (size_t)BB * NN * HH;
    _Float16* s16   = (_Float16*)(W + off); off += (size_t)BB * NN * HH / 2;
    _Float16* sel16 = (_Float16*)(W + off); off += (size_t)BB * NN * HH / 2;
    _Float16* Ws16  = (_Float16*)(W + off); off += (size_t)HH * HH / 2;
    _Float16* Wk16  = (_Float16*)(W + off); off += (size_t)HH * HH / 2;

    // pipeline (all on stream; veh_idx dependency resolved on-device)
    k_film<<<BB, HH, 0, stream>>>(weights, we1_w, we1_b, we2_w, we2_b,
                                  was_w, was_b, wad_w, wad_b, film);
    k_decoder<<<1, HH, 0, stream>>>(coord_w, coord_b, x0p, h0p, c0p,
                                    lstm_wih, lstm_whh, lstm_bih, lstm_bhh,
                                    gru_wih, gru_whh, gru_bih, gru_bhh,
                                    ro_w, ro_b, veh_in_w, veh_in_b,
                                    dec, out_ht, out_ct, out_hmem);
    k_sproj<<<BB * NN, HH, 0, stream>>>(stat_, static_w, static_b, film, s16);
    k_pack<<<HH, HH, 0, stream>>>(comb_w, cust_in_w, Ws16, Wk16);
    k_cv<<<BB * VV, HH, 0, stream>>>(trucks, drones, truck_w, truck_b,
                                     drone_w, drone_b, film, comb_w, comb_b, Cv);
    // A = s_t @ Ws^T : M=16384, Nc=256 -> 16384 tiles / 8 waves = 2048 blocks
    k_gemm_f16<<<(BB * NN / 16) * (HH / 16) / 8, 256, 0, stream>>>(s16, Ws16, Amat, HH);
    k_vehrepr<<<BB * VV, HH, 0, stream>>>(Amat, Cv, comb_g, comb_bb, vrep);
    k_vehattn<<<BB, HH, 0, stream>>>(dec, vrep, veh_in_w, veh_in_b,
                                     veh_out_w, veh_out_b, lnv_g, lnv_b, mask_v,
                                     vout, vidx, out_vehprobs, out_vehidx);
    k_sel<<<BB * NN, HH, 0, stream>>>(Amat, Cv, vidx, comb_g, comb_bb, sel16);
    // KH = sel @ Wk^T
    k_gemm_f16<<<(BB * NN / 16) * (HH / 16) / 8, 256, 0, stream>>>(sel16, Wk16, KHm, HH);
    k_custattn<<<BB, HH, 0, stream>>>(vout, KHm, cust_in_w, cust_in_b,
                                      mask_pw, vidx, out_custprobs);
}
